// RQVAE_85942295593528
// MI455X (gfx1250) — matrix operand order repaired and checked by reading, not compiled, not run on hardware
//
#include <hip/hip_runtime.h>
#include <cstdint>
#include <cstddef>

// ---------------- problem constants ----------------
constexpr int B_    = 16384;
constexpr int D_IN_ = 768;
constexpr int H_    = 1024;
constexpr int L_    = 128;
constexpr int K_    = 8192;
constexpr int E_    = 128;
constexpr int NL_   = 3;

// ---------------- WMMA types ----------------
typedef __attribute__((ext_vector_type(16))) __bf16 v16bf;
typedef __attribute__((ext_vector_type(8)))  float  v8f;

// native f32 -> bf16 convert (hardware v_cvt on gfx1250)
__device__ inline __bf16 f2bf(float x) { return (__bf16)x; }

__device__ inline v8f vzero8() {
    v8f z;
#pragma unroll
    for (int i = 0; i < 8; ++i) z[i] = 0.0f;
    return z;
}

// A fragment (16x32 bf16): lane holds row M=lane&15;
// lanes 0-15 carry K={0..15}, lanes 16-31 carry K={16..31}.
__device__ inline v16bf load_a_frag(const __bf16* base, int stride, int row,
                                    int lane, int kOff) {
    const __bf16* p = base + (size_t)(row + (lane & 15)) * stride + kOff
                    + ((lane & 16) ? 16 : 0);
    v16bf a;
#pragma unroll
    for (int q = 0; q < 4; ++q) { a[2*q]   = p[2*q];      a[2*q+1] = p[2*q+1]; }
#pragma unroll
    for (int q = 0; q < 4; ++q) { a[8+2*q] = p[8+2*q];   a[9+2*q] = p[8+2*q+1]; }
    return a;
}

// B fragment (32x16 bf16): staged transposed in LDS as [N][K] so each lane
// reads 16 contiguous K values. lane col N=lane&15; K base 0 / 16 per half.
__device__ inline v16bf load_b_frag(const __bf16* base, int stride, int col,
                                    int lane, int kOff) {
    const __bf16* p = base + (size_t)(col + (lane & 15)) * stride + kOff
                    + ((lane & 16) ? 16 : 0);
    v16bf b;
#pragma unroll
    for (int q = 0; q < 16; ++q) b[q] = p[q];
    return b;
}

__device__ inline v8f wmma_bf16(v16bf a, v16bf b, v8f c) {
    return __builtin_amdgcn_wmma_f32_16x16x32_bf16(
        /*neg_a=*/false, a, /*neg_b=*/false, b,
        /*c_mod=*/(short)0, c, /*reuse_a=*/false, /*reuse_b=*/false);
}

// ---------------- generic GEMM: C = act(A[MxK] * B[KxN] + bias) ------------
// block tile 128x128, BK=64, double-buffered LDS, 8 waves: 4 (M) x 2 (N),
// wave tile 32x64 -> 16 WMMA issues per barrier.
#define BM 128
#define BN 128
#define BKK 64

__global__ __launch_bounds__(256)
void gemm_bf16_wmma(const float* __restrict__ A, const float* __restrict__ Bm,
                    const float* __restrict__ bias, float* __restrict__ C,
                    int M, int N, int Kd, int relu) {
    __shared__ __bf16 As[2 * BM * BKK];   // [buf][row][k]     2 x 16 KB
    __shared__ __bf16 Bs[2 * BN * BKK];   // [buf][n][k] (T)   2 x 16 KB

    const int tid  = threadIdx.x;
    const int lane = tid & 31;
    const int wave = tid >> 5;
    const int wm   = wave & 3;        // 0..3 along M
    const int wn   = wave >> 2;       // 0..1 along N
    const int m0   = blockIdx.y * BM;
    const int n0   = blockIdx.x * BN;
    const int rowBase = wm * 32;
    const int colBase = wn * 64;

    // staging maps: A -> (row=tid/2, 32 consecutive k); B -> (k=tid/4, 32 n)
    const int rA = tid >> 1, kA = (tid & 1) * 32;
    const int kB = tid >> 2, nB = (tid & 3) * 32;

    v8f acc[2][4];
#pragma unroll
    for (int i = 0; i < 2; ++i)
#pragma unroll
        for (int j = 0; j < 4; ++j) acc[i][j] = vzero8();

    float4 ar[8], br[8];
    auto gload = [&](int kt) {
        const float* ga = A + (size_t)(m0 + rA) * Kd + kt + kA;
#pragma unroll
        for (int j = 0; j < 8; ++j) ar[j] = *(const float4*)(ga + 4 * j);
        const float* gb = Bm + (size_t)(kt + kB) * N + n0 + nB;
#pragma unroll
        for (int j = 0; j < 8; ++j) br[j] = *(const float4*)(gb + 4 * j);
    };
    auto sstore = [&](int buf) {
        __bf16* da = As + buf * (BM * BKK) + rA * BKK + kA;
#pragma unroll
        for (int j = 0; j < 8; ++j) {
            da[4*j+0] = f2bf(ar[j].x); da[4*j+1] = f2bf(ar[j].y);
            da[4*j+2] = f2bf(ar[j].z); da[4*j+3] = f2bf(ar[j].w);
        }
        __bf16* db = Bs + buf * (BN * BKK);
#pragma unroll
        for (int j = 0; j < 8; ++j) {
            db[(nB + 4*j + 0) * BKK + kB] = f2bf(br[j].x);
            db[(nB + 4*j + 1) * BKK + kB] = f2bf(br[j].y);
            db[(nB + 4*j + 2) * BKK + kB] = f2bf(br[j].z);
            db[(nB + 4*j + 3) * BKK + kB] = f2bf(br[j].w);
        }
    };

    const int nStages = Kd / BKK;
    gload(0);
    sstore(0);
    __syncthreads();

    for (int t = 0; t < nStages; ++t) {
        const int cur = t & 1;
        if (t + 1 < nStages) gload((t + 1) * BKK);   // in flight during WMMA

        const __bf16* Ab = As + cur * (BM * BKK);
        const __bf16* Bb = Bs + cur * (BN * BKK);
#pragma unroll
        for (int ks = 0; ks < BKK; ks += 32) {
            v16bf af[2], bfv[4];
#pragma unroll
            for (int i = 0; i < 2; ++i)
                af[i] = load_a_frag(Ab, BKK, rowBase + i * 16, lane, ks);
#pragma unroll
            for (int j = 0; j < 4; ++j)
                bfv[j] = load_b_frag(Bb, BKK, colBase + j * 16, lane, ks);
#pragma unroll
            for (int i = 0; i < 2; ++i)
#pragma unroll
                for (int j = 0; j < 4; ++j)
                    acc[i][j] = wmma_bf16(af[i], bfv[j], acc[i][j]);
        }
        if (t + 1 < nStages) sstore(1 - cur);
        __syncthreads();
    }

    // epilogue: D layout — N=lane&15, M = v + 8*(lane>=16)
    const int nl   = lane & 15;
    const int half = lane >> 4;
#pragma unroll
    for (int i = 0; i < 2; ++i)
#pragma unroll
        for (int j = 0; j < 4; ++j) {
            int gn = n0 + wn * 64 + j * 16 + nl;
            float bv = bias[gn];
#pragma unroll
            for (int v = 0; v < 8; ++v) {
                int gm = m0 + wm * 32 + i * 16 + half * 8 + v;
                float val = acc[i][j][v] + bv;
                if (relu) val = fmaxf(val, 0.0f);
                C[(size_t)gm * N + gn] = val;
            }
        }
}

// ---------------- fused codebook distance + argmin --------------------------
// block: 64 rows of f; sweep all K codes in chunks of 64, double-buffered.
// argmin_n(|f|^2+|cb_n|^2-2 f.cb_n) == argmin_n(|cb_n|^2 - 2 f.cb_n)
__global__ __launch_bounds__(256)
void rq_argmin(const float* __restrict__ f, const float* __restrict__ cb,
               const float* __restrict__ cbn, int* __restrict__ idx,
               int Kcodes) {
    __shared__ __bf16 Fs[64 * E_];        // 16 KB  [row][e]
    __shared__ __bf16 Cs[2 * 64 * E_];    // 32 KB  [buf][code][e]
    __shared__ float  Sc[64 * 64];        // 16 KB  score tile

    const int tid  = threadIdx.x;
    const int lane = tid & 31;
    const int wave = tid >> 5;
    const int rowBase = (wave & 3) * 16;   // 4 waves along M
    const int colBase = (wave >> 2) * 32;  // 2 waves along N, 2 subtiles each
    const int row0 = blockIdx.x * 64;

    const int rT = tid >> 2;               // 0..63
    const int e0 = (tid & 3) * 32;

    {   // stage f rows once
        const float* g = f + (size_t)(row0 + rT) * E_ + e0;
        __bf16* d = Fs + rT * E_ + e0;
#pragma unroll
        for (int j = 0; j < 32; j += 4) {
            float4 v = *(const float4*)(g + j);
            d[j+0] = f2bf(v.x); d[j+1] = f2bf(v.y);
            d[j+2] = f2bf(v.z); d[j+3] = f2bf(v.w);
        }
    }

    float4 cr[8];
    auto gloadC = [&](int c0) {
        const float* g = cb + (size_t)(c0 + rT) * E_ + e0;
#pragma unroll
        for (int j = 0; j < 8; ++j) cr[j] = *(const float4*)(g + 4 * j);
    };
    auto sstoreC = [&](int buf) {
        __bf16* d = Cs + buf * (64 * E_) + rT * E_ + e0;
#pragma unroll
        for (int j = 0; j < 8; ++j) {
            d[4*j+0] = f2bf(cr[j].x); d[4*j+1] = f2bf(cr[j].y);
            d[4*j+2] = f2bf(cr[j].z); d[4*j+3] = f2bf(cr[j].w);
        }
    };

    gloadC(0);
    sstoreC(0);
    __syncthreads();

    float bestV = 3.402823466e+38f;
    int   bestI = 0;
    const int nChunks = Kcodes / 64;

    for (int t = 0; t < nChunks; ++t) {
        const int cur = t & 1;
        const int c0  = t * 64;
        if (t + 1 < nChunks) gloadC(c0 + 64);   // in flight during WMMA

        const __bf16* Cb = Cs + cur * (64 * E_);
        v8f acc[2] = { vzero8(), vzero8() };
#pragma unroll
        for (int ks = 0; ks < E_; ks += 32) {
            v16bf a = load_a_frag(Fs, E_, rowBase, lane, ks);
#pragma unroll
            for (int j = 0; j < 2; ++j) {
                v16bf b = load_b_frag(Cb, E_, colBase + j * 16, lane, ks);
                acc[j] = wmma_bf16(a, b, acc[j]);
            }
        }
        const int nl   = lane & 15;
        const int half = lane >> 4;
#pragma unroll
        for (int j = 0; j < 2; ++j)
#pragma unroll
            for (int v = 0; v < 8; ++v)
                Sc[(rowBase + half * 8 + v) * 64 + (colBase + j * 16 + nl)]
                    = acc[j][v];
        if (t + 1 < nChunks) sstoreC(1 - cur);
        __syncthreads();

        if (tid < 64) {   // row-owner scan, first-min semantics
            const float* sr = Sc + tid * 64;
            for (int n = 0; n < 64; ++n) {
                float val = cbn[c0 + n] - 2.0f * sr[n];
                if (val < bestV) { bestV = val; bestI = c0 + n; }
            }
        }
        __syncthreads();   // scanners done before Sc overwritten next iter
    }
    if (tid < 64) idx[row0 + tid] = bestI;
}

// ---------------- small kernels ---------------------------------------------
__global__ __launch_bounds__(256)
void cbnorm_kernel(const float* __restrict__ cb, float* __restrict__ out,
                   int total) {
    int i = blockIdx.x * 256 + threadIdx.x;
    if (i >= total) return;
    const float* p = cb + (size_t)i * E_;
    float s = 0.0f;
#pragma unroll 4
    for (int e = 0; e < E_; e += 4) {
        float4 v = *(const float4*)(p + e);
        s += v.x*v.x + v.y*v.y + v.z*v.z + v.w*v.w;
    }
    out[i] = s;
}

// one wave per 128-wide row, torch-style LayerNorm eps=1e-5
__global__ __launch_bounds__(256)
void ln_kernel(const float* __restrict__ x, const float* __restrict__ g,
               const float* __restrict__ b, float* __restrict__ y) {
    int lane = threadIdx.x & 31;
    int row  = blockIdx.x * 8 + (threadIdx.x >> 5);
    float4 v = ((const float4*)(x + (size_t)row * L_))[lane];
    float s = v.x + v.y + v.z + v.w;
#pragma unroll
    for (int m = 16; m; m >>= 1) s += __shfl_xor(s, m, 32);
    float mean = s * (1.0f / L_);
    float d0 = v.x - mean, d1 = v.y - mean, d2 = v.z - mean, d3 = v.w - mean;
    float sq = d0*d0 + d1*d1 + d2*d2 + d3*d3;
#pragma unroll
    for (int m = 16; m; m >>= 1) sq += __shfl_xor(sq, m, 32);
    float inv = rsqrtf(sq * (1.0f / L_) + 1e-5f);
    float4 gg = ((const float4*)g)[lane];
    float4 bb = ((const float4*)b)[lane];
    float4 o;
    o.x = d0 * inv * gg.x + bb.x;  o.y = d1 * inv * gg.y + bb.y;
    o.z = d2 * inv * gg.z + bb.z;  o.w = d3 * inv * gg.w + bb.w;
    ((float4*)(y + (size_t)row * L_))[lane] = o;
}

// z = mu + eps*exp(0.5*lv); cur = z; quantized = 0; loss = 0
__global__ __launch_bounds__(256)
void reparam_kernel(const float* __restrict__ mu, const float* __restrict__ lv,
                    const float* __restrict__ eps, float* __restrict__ cur,
                    float* __restrict__ quant, float* __restrict__ loss,
                    int n) {
    int i = blockIdx.x * 256 + threadIdx.x;
    if (i < n) {
        float z = mu[i] + eps[i] * expf(0.5f * lv[i]);
        cur[i] = z;
        quant[i] = 0.0f;
    }
    if (i == 0) *loss = 0.0f;
}

__global__ __launch_bounds__(256)
void gather_kernel(const float* __restrict__ cb, const int* __restrict__ idx,
                   float* __restrict__ fq, int n) {
    int i = blockIdx.x * 256 + threadIdx.x;
    if (i >= n) return;
    int b = i >> 7, e = i & 127;
    fq[i] = cb[(size_t)idx[b] * E_ + e];
}

// quantized += q; cur -= q; loss += scale * sum((q - nr)^2)
__global__ __launch_bounds__(256)
void update_kernel(const float* __restrict__ q, const float* __restrict__ nr,
                   float* __restrict__ cur, float* __restrict__ quant,
                   float* __restrict__ loss, float scale, int n) {
    __shared__ float red[256];
    int tid = threadIdx.x;
    int i = blockIdx.x * 256 + tid;
    float d = 0.0f;
    if (i < n) {
        float qv = q[i];
        float diff = qv - nr[i];
        d = diff * diff;
        quant[i] += qv;
        cur[i]   -= qv;
    }
    red[tid] = d;
    __syncthreads();
#pragma unroll
    for (int s = 128; s > 0; s >>= 1) {
        if (tid < s) red[tid] += red[tid + s];
        __syncthreads();
    }
    if (tid == 0) atomicAdd(loss, red[0] * scale);
}

// ---------------- host orchestration ----------------------------------------
extern "C" void kernel_launch(void* const* d_in, const int* in_sizes, int n_in,
                              void* d_out, int out_size, void* d_ws,
                              size_t ws_size, hipStream_t stream) {
    (void)in_sizes; (void)n_in; (void)out_size; (void)ws_size;

    const float* x        = (const float*)d_in[0];
    const float* eps      = (const float*)d_in[1];
    const float* enc_w1   = (const float*)d_in[2];
    const float* enc_b1   = (const float*)d_in[3];
    const float* enc_w2   = (const float*)d_in[4];
    const float* enc_b2   = (const float*)d_in[5];
    const float* enc_w3   = (const float*)d_in[6];
    const float* enc_b3   = (const float*)d_in[7];
    const float* mu_w     = (const float*)d_in[8];
    const float* mu_b     = (const float*)d_in[9];
    const float* var_w    = (const float*)d_in[10];
    const float* var_b    = (const float*)d_in[11];
    const float* dec_w1   = (const float*)d_in[12];
    const float* dec_b1   = (const float*)d_in[13];
    const float* dec_w2   = (const float*)d_in[14];
    const float* dec_b2   = (const float*)d_in[15];
    const float* dec_w3   = (const float*)d_in[16];
    const float* dec_b3   = (const float*)d_in[17];
    const float* ln_g     = (const float*)d_in[18];
    const float* ln_b     = (const float*)d_in[19];
    const float* q_in_w   = (const float*)d_in[20];
    const float* q_in_b   = (const float*)d_in[21];
    const float* codebook = (const float*)d_in[22];
    const float* q_out_w  = (const float*)d_in[23];
    const float* q_out_b  = (const float*)d_in[24];

    float* out    = (float*)d_out;
    float* recon  = out;                              // [B, D_IN]
    float* mu_out = out + (size_t)B_ * D_IN_;         // [B, L]
    float* lv_out = mu_out + (size_t)B_ * L_;         // [B, L]
    float* loss   = lv_out + (size_t)B_ * L_;         // [1]

    float* ws    = (float*)d_ws;
    float* bufA  = ws;  ws += (size_t)B_ * H_;
    float* bufB  = ws;  ws += (size_t)B_ * H_;
    float* cur   = ws;  ws += (size_t)B_ * L_;
    float* quant = ws;  ws += (size_t)B_ * L_;
    float* nr    = ws;  ws += (size_t)B_ * L_;
    float* fbuf  = ws;  ws += (size_t)B_ * L_;
    float* fq    = ws;  ws += (size_t)B_ * L_;
    float* qbuf  = ws;  ws += (size_t)B_ * L_;
    float* cbn   = ws;  ws += (size_t)NL_ * K_;
    int*   idxb  = (int*)ws;

    const dim3 blk(256);
    auto gemm = [&](const float* A, const float* Bm, const float* bias,
                    float* C, int M, int N, int Kd, int relu) {
        dim3 grid(N / BN, M / BM);
        gemm_bf16_wmma<<<grid, blk, 0, stream>>>(A, Bm, bias, C, M, N, Kd, relu);
    };

    // encoder
    gemm(x,    enc_w1, enc_b1, bufA, B_, H_, D_IN_, 1);
    gemm(bufA, enc_w2, enc_b2, bufB, B_, H_, H_,    1);
    gemm(bufB, enc_w3, enc_b3, bufA, B_, H_, H_,    0);
    gemm(bufA, mu_w,   mu_b,   mu_out, B_, L_, H_,  0);
    gemm(bufA, var_w,  var_b,  lv_out, B_, L_, H_,  0);

    // codebook norms + reparameterization
    cbnorm_kernel<<<dim3((NL_ * K_) / 256), blk, 0, stream>>>(codebook, cbn,
                                                              NL_ * K_);
    reparam_kernel<<<dim3((B_ * L_) / 256), blk, 0, stream>>>(
        mu_out, lv_out, eps, cur, quant, loss, B_ * L_);

    // residual quantization
    const float lscale = 1.25f / (float)(B_ * L_);   // (1 + CC) / (B*L)
    for (int l = 0; l < NL_; ++l) {
        ln_kernel<<<dim3(B_ / 8), blk, 0, stream>>>(
            cur, ln_g + (size_t)l * L_, ln_b + (size_t)l * L_, nr);
        gemm(nr, q_in_w + (size_t)l * L_ * E_, q_in_b + (size_t)l * E_,
             fbuf, B_, E_, L_, 0);
        rq_argmin<<<dim3(B_ / 64), blk, 0, stream>>>(
            fbuf, codebook + (size_t)l * K_ * E_, cbn + (size_t)l * K_,
            idxb, K_);
        gather_kernel<<<dim3((B_ * E_) / 256), blk, 0, stream>>>(
            codebook + (size_t)l * K_ * E_, idxb, fq, B_ * E_);
        gemm(fq, q_out_w + (size_t)l * E_ * L_, q_out_b + (size_t)l * L_,
             qbuf, B_, L_, E_, 0);
        update_kernel<<<dim3((B_ * L_) / 256), blk, 0, stream>>>(
            qbuf, nr, cur, quant, loss, lscale, B_ * L_);
    }

    // decoder
    gemm(quant, dec_w1, dec_b1, bufA, B_, H_,    L_, 1);
    gemm(bufA,  dec_w2, dec_b2, bufB, B_, H_,    H_, 1);
    gemm(bufB,  dec_w3, dec_b3, recon, B_, D_IN_, H_, 0);
}
